// Attention_86268713107668
// MI455X (gfx1250) — compile-verified, matrix operand order checked by reading, not run on hardware
//
#include <hip/hip_runtime.h>
#include <cstdint>
#include <cstddef>

// ---------------------------------------------------------------------------
// Types
// ---------------------------------------------------------------------------
typedef __bf16 bf16_t;
typedef __attribute__((ext_vector_type(16))) __bf16 v16bf;
typedef __attribute__((ext_vector_type(4)))  __bf16 v4bf;
typedef __attribute__((ext_vector_type(8)))  float  v8f;
typedef int v4i_vs __attribute__((vector_size(16)));   // matches builtin proto

struct __attribute__((aligned(16))) U4 { unsigned int a, b, c, d; };
struct __attribute__((aligned(16))) F4 { float a, b, c, d; };

#define WMMA_BF16(A_, B_, C_) \
  __builtin_amdgcn_wmma_f32_16x16x32_bf16(false, (A_), false, (B_), (short)0, (C_), false, false)

// --- CDNA5 async global->LDS path (ASYNCcnt), with sync fallback -----------
#if defined(__gfx1250__) && __has_builtin(__builtin_amdgcn_global_load_async_to_lds_b128)
#define USE_ASYNC_LDS 1
#else
#define USE_ASYNC_LDS 0
#endif

// Builtin signature: (v4i addrspace(1)*, v4i addrspace(3)*, imm offset, imm cpol).
// Generic->AS1: global generic addresses are the 64-bit VA.
// Generic->AS3: low 32 bits of a generic LDS address are the LDS byte offset.
#define TO_GLOBAL_V4I(p) \
  ((__attribute__((address_space(1))) v4i_vs*)(uintptr_t)(p))
#define TO_LDS_V4I(p) \
  ((__attribute__((address_space(3))) v4i_vs*)(unsigned int)(uintptr_t)(p))

__device__ __forceinline__ void cp_b128(const bf16_t* g, bf16_t* l)
{
#if USE_ASYNC_LDS
  __builtin_amdgcn_global_load_async_to_lds_b128(TO_GLOBAL_V4I(g), TO_LDS_V4I(l), 0, 0);
#else
  *(U4*)l = *(const U4*)g;
#endif
}

__device__ __forceinline__ void async_wait0()
{
#if USE_ASYNC_LDS
#if __has_builtin(__builtin_amdgcn_s_wait_asynccnt)
  __builtin_amdgcn_s_wait_asynccnt(0);
#else
  asm volatile("s_wait_asynccnt 0" ::: "memory");
#endif
#endif
}

// Load one 16x32 bf16 WMMA A-operand (or the mirrored Bt operand) fragment.
// ISA layout (16-bit A 16x32): lanes 0-15 -> row = lane, k in {0..7,16..23};
// lanes 16-31 -> row = lane-16, k in {8..15,24..31}.  Two b128 loads per lane.
__device__ __forceinline__ v16bf frag_ld(const bf16_t* p, int row0, int stride,
                                         int kc0, int lane)
{
  int r  = row0 + (lane & 15);
  int kh = ((lane >> 4) << 3) + kc0;
  const bf16_t* q = p + (size_t)r * stride + kh;
  union { U4 u[2]; v16bf v; } uu;
  uu.u[0] = *(const U4*)(q);
  uu.u[1] = *(const U4*)(q + 16);
  return uu.v;
}

// ---------------------------------------------------------------------------
// Prep kernels
// ---------------------------------------------------------------------------
__global__ __launch_bounds__(256) void conv_in_kernel(
    const float* __restrict__ X, const unsigned char* __restrict__ mask,
    bf16_t* __restrict__ out)
{
  int idx = blockIdx.x * 256 + threadIdx.x;   // 1M threads, 4 elems each
  int row = idx >> 8;                         // DIM/4 = 256 vec4 per row
  F4 v = *(const F4*)(X + (size_t)idx * 4);
  bool m = mask[row] != 0;
  v4bf o;
  o[0] = m ? (bf16_t)0.f : (bf16_t)v.a;
  o[1] = m ? (bf16_t)0.f : (bf16_t)v.b;
  o[2] = m ? (bf16_t)0.f : (bf16_t)v.c;
  o[3] = m ? (bf16_t)0.f : (bf16_t)v.d;
  *(v4bf*)(out + (size_t)idx * 4) = o;
}

__global__ __launch_bounds__(256) void conv_w_kernel(
    const float* __restrict__ W, bf16_t* __restrict__ out)
{
  int idx = blockIdx.x * 256 + threadIdx.x;   // 256K threads, 4 elems each
  F4 v = *(const F4*)(W + (size_t)idx * 4);
  v4bf o;
  o[0] = (bf16_t)v.a; o[1] = (bf16_t)v.b; o[2] = (bf16_t)v.c; o[3] = (bf16_t)v.d;
  *(v4bf*)(out + (size_t)idx * 4) = o;
}

__global__ __launch_bounds__(256) void bias_k_kernel(
    const unsigned char* __restrict__ maskK, float* __restrict__ biasK)
{
  int idx = blockIdx.x * 256 + threadIdx.x;   // B*Nk = 4096
  biasK[idx] = (maskK[idx] != 0) ? -1.0e30f : 0.0f;
}

// ---------------------------------------------------------------------------
// GEMM: C[M,N] = A[M,K] (bf16, row-major) x Bt[N,K]^T (bf16, row-major)
// 256x128 block tile, 8 wave32 waves of 64x64, BK=32, double-buffered LDS
// filled via async global->LDS DMA.  16 WMMAs per 8 fragment loads per wave
// per K-step.
// ---------------------------------------------------------------------------
__global__ __launch_bounds__(256) void gemm_bf16_tn(
    const bf16_t* __restrict__ A, const bf16_t* __restrict__ Bt,
    float* __restrict__ C, int M, int N, int K)
{
  constexpr int BM = 256, BN = 128, BK = 32;
  __shared__ __attribute__((aligned(16))) bf16_t sA[2][BM * BK];  // 16KB x2
  __shared__ __attribute__((aligned(16))) bf16_t sB[2][BN * BK];  //  8KB x2
  const int tid = threadIdx.x, lane = tid & 31, wid = tid >> 5;
  const int bm = blockIdx.x * BM, bn = blockIdx.y * BN;
  const int wm = (wid >> 1) * 64, wn = (wid & 1) * 64;
  (void)M;

  auto stage = [&](int kb, int buf) {
    const bf16_t* Ag = A  + (size_t)bm * K + (size_t)kb * BK;
    const bf16_t* Bg = Bt + (size_t)bn * K + (size_t)kb * BK;
#pragma unroll
    for (int c = 0; c < 4; ++c) {              // A: 1024 chunks of 8 bf16
      int qi  = c * 256 + tid;
      int row = qi >> 2;
      int col = (qi & 3) * 8;
      cp_b128(Ag + (size_t)row * K + col, &sA[buf][row * BK + col]);
#if !USE_ASYNC_LDS
      __builtin_prefetch((const void*)(Ag + (size_t)row * K + col + 2 * BK), 0, 0);
#endif
    }
#pragma unroll
    for (int c = 0; c < 2; ++c) {              // B: 512 chunks of 8 bf16
      int qi  = c * 256 + tid;
      int row = qi >> 2;
      int col = (qi & 3) * 8;
      cp_b128(Bg + (size_t)row * K + col, &sB[buf][row * BK + col]);
#if !USE_ASYNC_LDS
      __builtin_prefetch((const void*)(Bg + (size_t)row * K + col + 2 * BK), 0, 0);
#endif
    }
  };

  v8f zf = {0.f, 0.f, 0.f, 0.f, 0.f, 0.f, 0.f, 0.f};
  v8f acc[4][4];
#pragma unroll
  for (int i = 0; i < 4; ++i)
#pragma unroll
    for (int j = 0; j < 4; ++j) acc[i][j] = zf;

  const int nk = K / BK;
  stage(0, 0);
  async_wait0();
  __syncthreads();
  for (int kb = 0; kb < nk; ++kb) {
    int buf = kb & 1;
    if (kb + 1 < nk) stage(kb + 1, buf ^ 1);
    v16bf bfr[4];
#pragma unroll
    for (int j = 0; j < 4; ++j)
      bfr[j] = frag_ld(sB[buf], wn + j * 16, BK, 0, lane);
#pragma unroll
    for (int i = 0; i < 4; ++i) {
      v16bf af = frag_ld(sA[buf], wm + i * 16, BK, 0, lane);
#pragma unroll
      for (int j = 0; j < 4; ++j)
        acc[i][j] = WMMA_BF16(af, bfr[j], acc[i][j]);
    }
    async_wait0();       // next tile's DMA done before the barrier releases it
    __syncthreads();
  }

  // C layout: VGPR r, lanes 0-15 -> (row r, col lane); lanes 16-31 -> (row r+8).
#pragma unroll
  for (int i = 0; i < 4; ++i)
#pragma unroll
    for (int j = 0; j < 4; ++j)
#pragma unroll
      for (int r = 0; r < 8; ++r) {
        int m = bm + wm + i * 16 + r + 8 * (lane >> 4);
        int n = bn + wn + j * 16 + (lane & 15);
        C[(size_t)m * N + n] = acc[i][j][r];
      }
}

// ---------------------------------------------------------------------------
// Repack kernels: head split / transpose into attention layouts (bf16).
// ---------------------------------------------------------------------------
// out[((b*16+h)*1024 + n)*64 + d] = P[(b*1024+n)*1024 + h*64 + d]
__global__ __launch_bounds__(256) void repack_qk_kernel(
    const float* __restrict__ P, bf16_t* __restrict__ out)
{
  size_t idx = (size_t)blockIdx.x * 256 + threadIdx.x;  // 1M, 4 d-elems each
  size_t t = idx;
  int d4 = (int)(t & 15);  t >>= 4;
  int n  = (int)(t & 1023); t >>= 10;
  int h  = (int)(t & 15);
  int b  = (int)(t >> 4);
  const F4 v = *(const F4*)(P + ((size_t)(b * 1024 + n) * 1024 + h * 64 + d4 * 4));
  v4bf o;
  o[0] = (bf16_t)v.a; o[1] = (bf16_t)v.b; o[2] = (bf16_t)v.c; o[3] = (bf16_t)v.d;
  *(v4bf*)(out + idx * 4) = o;
}

// vT[((b*16+h)*64 + d)*1024 + k] = Vp[(b*1024+k)*1024 + h*64 + d]
__global__ __launch_bounds__(256) void repack_v_kernel(
    const float* __restrict__ Vp, bf16_t* __restrict__ vT)
{
  size_t idx = (size_t)blockIdx.x * 256 + threadIdx.x;  // 1M, 4 k-elems each
  size_t t = idx;
  int k4 = (int)(t & 255); t >>= 8;
  int d  = (int)(t & 63);  t >>= 6;
  int h  = (int)(t & 15);
  int b  = (int)(t >> 4);
  int k0 = k4 * 4;
  v4bf o;
#pragma unroll
  for (int j = 0; j < 4; ++j)
    o[j] = (bf16_t)Vp[(size_t)(b * 1024 + k0 + j) * 1024 + h * 64 + d];
  *(v4bf*)(vT + idx * 4) = o;
}

// ---------------------------------------------------------------------------
// Flash attention: per (b,h), 128 queries per block, 8 waves x 16 queries.
// S = Q.K^T via WMMA; online softmax (row stats replicated per half-wave);
// P round-trips per-wave LDS to WMMA A-layout; O += P.V^T via WMMA.
// K / V^T tiles double-buffered in LDS via async global->LDS DMA.
// ---------------------------------------------------------------------------
__global__ __launch_bounds__(256) void flash_attn_kernel(
    const bf16_t* __restrict__ qbf,   // [B,H,N,64]
    const bf16_t* __restrict__ kbf,   // [B,H,N,64]
    const bf16_t* __restrict__ vTbf,  // [B,H,64,N]
    const float* __restrict__ biasK,  // [B,N]
    float* __restrict__ Oatt)         // [B,N,DIM]
{
  constexpr int N = 1024, DH = 64, HN = 16;
  __shared__ __attribute__((aligned(16))) bf16_t sK[2][32 * DH];
  __shared__ __attribute__((aligned(16))) bf16_t sV[2][DH * 32];
  __shared__ __attribute__((aligned(16))) bf16_t sP[8][16 * 32];
  const int tid = threadIdx.x, lane = tid & 31, wid = tid >> 5;
  const int b = blockIdx.z, h = blockIdx.y, qb = blockIdx.x;
  const size_t hoff = (size_t)(b * HN + h) * N * DH;
  const bf16_t* Qh = qbf + hoff;
  const bf16_t* Kh = kbf + hoff;
  const bf16_t* Vh = vTbf + hoff;
  const float*  brow = biasK + (size_t)b * N;
  const int q0 = qb * 128 + wid * 16;

  v16bf aq0 = frag_ld(Qh + (size_t)q0 * DH, 0, DH, 0,  lane);
  v16bf aq1 = frag_ld(Qh + (size_t)q0 * DH, 0, DH, 32, lane);

  v8f zf = {0.f, 0.f, 0.f, 0.f, 0.f, 0.f, 0.f, 0.f};
  v8f accO[4];
#pragma unroll
  for (int dt = 0; dt < 4; ++dt) accO[dt] = zf;
  float mrun[8], lrun[8];
#pragma unroll
  for (int r = 0; r < 8; ++r) { mrun[r] = -1.0e30f; lrun[r] = 0.f; }

  auto stageKV = [&](int kb, int buf) {
    { int row = tid >> 3, ch = (tid & 7) * 8;        // K tile: 32 x 64
      cp_b128(Kh + (size_t)(kb * 32 + row) * DH + ch, &sK[buf][row * DH + ch]); }
    { int row = tid >> 2, ch = (tid & 3) * 8;        // V^T tile: 64 x 32
      cp_b128(Vh + (size_t)row * N + kb * 32 + ch, &sV[buf][row * 32 + ch]); }
  };

  stageKV(0, 0);
  async_wait0();
  __syncthreads();
  for (int kb = 0; kb < N / 32; ++kb) {
    int buf = kb & 1;
    if (kb + 1 < N / 32) stageKV(kb + 1, buf ^ 1);

    v8f s0 = zf, s1 = zf;
    v16bf bk;
    bk = frag_ld(sK[buf], 0,  DH, 0,  lane); s0 = WMMA_BF16(aq0, bk, s0);
    bk = frag_ld(sK[buf], 0,  DH, 32, lane); s0 = WMMA_BF16(aq1, bk, s0);
    bk = frag_ld(sK[buf], 16, DH, 0,  lane); s1 = WMMA_BF16(aq0, bk, s1);
    bk = frag_ld(sK[buf], 16, DH, 32, lane); s1 = WMMA_BF16(aq1, bk, s1);

    float bj0 = brow[kb * 32 + (lane & 15)];
    float bj1 = brow[kb * 32 + 16 + (lane & 15)];

    float p0[8], p1[8];
#pragma unroll
    for (int r = 0; r < 8; ++r) {
      float a = s0[r] * 0.03125f + bj0;       // 1/sqrt(1024)
      float c = s1[r] * 0.03125f + bj1;
      float t = fmaxf(a, c);
      t = fmaxf(t, __shfl_xor(t, 1, 16));
      t = fmaxf(t, __shfl_xor(t, 2, 16));
      t = fmaxf(t, __shfl_xor(t, 4, 16));
      t = fmaxf(t, __shfl_xor(t, 8, 16));
      float mn   = fmaxf(mrun[r], t);
      float corr = __expf(mrun[r] - mn);
      float e0 = __expf(a - mn);
      float e1 = __expf(c - mn);
      float rs = e0 + e1;
      rs += __shfl_xor(rs, 1, 16);
      rs += __shfl_xor(rs, 2, 16);
      rs += __shfl_xor(rs, 4, 16);
      rs += __shfl_xor(rs, 8, 16);
      lrun[r] = lrun[r] * corr + rs;
      mrun[r] = mn;
#pragma unroll
      for (int dt = 0; dt < 4; ++dt) accO[dt][r] *= corr;
      p0[r] = e0; p1[r] = e1;
    }

    // C-layout -> A-layout via per-wave LDS round trip
    bf16_t* pp = sP[wid];
    int coln = lane & 15, rb = (lane >> 4) * 8;
#pragma unroll
    for (int r = 0; r < 8; ++r) {
      pp[(rb + r) * 32 + coln]      = (bf16_t)p0[r];
      pp[(rb + r) * 32 + 16 + coln] = (bf16_t)p1[r];
    }
    asm volatile("s_wait_dscnt 0" ::: "memory");   // same-wave LDS RAW fence
    v16bf ap = frag_ld(pp, 0, 32, 0, lane);
#pragma unroll
    for (int dt = 0; dt < 4; ++dt) {
      v16bf bv = frag_ld(sV[buf], dt * 16, 32, 0, lane);
      accO[dt] = WMMA_BF16(ap, bv, accO[dt]);
    }
    async_wait0();
    __syncthreads();
  }

#pragma unroll
  for (int dt = 0; dt < 4; ++dt)
#pragma unroll
    for (int r = 0; r < 8; ++r) {
      int q = q0 + r + 8 * (lane >> 4);
      int d = dt * 16 + (lane & 15);
      float l = lrun[r];
      float val = (l > 0.f) ? accO[dt][r] / l : 0.f;
      Oatt[((size_t)b * N + q) * (HN * DH) + h * DH + d] = val;
    }
}

// ---------------------------------------------------------------------------
// LayerNorm kernels (one 256-thread block per row of 1024)
// ---------------------------------------------------------------------------
__device__ __forceinline__ float block_sum256(float v)
{
  __shared__ float red[8];
  int lane = threadIdx.x & 31, wid = threadIdx.x >> 5;
#pragma unroll
  for (int off = 16; off > 0; off >>= 1) v += __shfl_xor(v, off, 32);
  if (lane == 0) red[wid] = v;
  __syncthreads();
  float s = 0.f;
#pragma unroll
  for (int i = 0; i < 8; ++i) s += red[i];
  __syncthreads();
  return s;
}

__global__ __launch_bounds__(256) void ln1_kernel(
    const float* __restrict__ Qp, const float* __restrict__ Oatt,
    const float* __restrict__ g, const float* __restrict__ bb,
    const unsigned char* __restrict__ maskQ,
    float* __restrict__ Of, bf16_t* __restrict__ Obf)
{
  int row = blockIdx.x, tid = threadIdx.x;
  const float* x1 = Qp   + (size_t)row * 1024;
  const float* x2 = Oatt + (size_t)row * 1024;
  float v[4]; float s = 0.f;
#pragma unroll
  for (int i = 0; i < 4; ++i) { int c = tid + i * 256; v[i] = x1[c] + x2[c]; s += v[i]; }
  float mean = block_sum256(s) * (1.0f / 1024.0f);
  float vr = 0.f;
#pragma unroll
  for (int i = 0; i < 4; ++i) { float d = v[i] - mean; vr += d * d; }
  float var = block_sum256(vr) * (1.0f / 1024.0f);
  float rstd = rsqrtf(var + 1e-5f);
  bool msk = maskQ[row] != 0;
#pragma unroll
  for (int i = 0; i < 4; ++i) {
    int c = tid + i * 256;
    float y = msk ? 0.f : (v[i] - mean) * rstd * g[c] + bb[c];
    Of[(size_t)row * 1024 + c]  = y;
    Obf[(size_t)row * 1024 + c] = (bf16_t)y;
  }
}

__global__ __launch_bounds__(256) void ln2_kernel(
    const float* __restrict__ Of, const float* __restrict__ Hf,
    const float* __restrict__ g, const float* __restrict__ bb,
    const unsigned char* __restrict__ maskQ, float* __restrict__ out)
{
  int row = blockIdx.x, tid = threadIdx.x;
  float v[4]; float s = 0.f;
#pragma unroll
  for (int i = 0; i < 4; ++i) {
    int c = tid + i * 256;
    float o  = Of[(size_t)row * 1024 + c];
    float hh = Hf[(size_t)row * 1024 + c];
    float ge = 0.5f * hh * (1.0f + erff(hh * 0.70710678118654752f));  // exact GELU
    v[i] = o + ge; s += v[i];
  }
  float mean = block_sum256(s) * (1.0f / 1024.0f);
  float vr = 0.f;
#pragma unroll
  for (int i = 0; i < 4; ++i) { float d = v[i] - mean; vr += d * d; }
  float var = block_sum256(vr) * (1.0f / 1024.0f);
  float rstd = rsqrtf(var + 1e-5f);
  bool msk = maskQ[row] != 0;
#pragma unroll
  for (int i = 0; i < 4; ++i) {
    int c = tid + i * 256;
    out[(size_t)row * 1024 + c] = msk ? 0.f : (v[i] - mean) * rstd * g[c] + bb[c];
  }
}

// ---------------------------------------------------------------------------
// Host orchestration
// ---------------------------------------------------------------------------
extern "C" void kernel_launch(void* const* d_in, const int* in_sizes, int n_in,
                              void* d_out, int out_size, void* d_ws, size_t ws_size,
                              hipStream_t stream)
{
  (void)in_sizes; (void)n_in; (void)out_size; (void)ws_size;
  const float* Q     = (const float*)d_in[0];
  const float* K     = (const float*)d_in[1];
  const float* V     = (const float*)d_in[2];
  const float* Wq    = (const float*)d_in[3];
  const float* Wk    = (const float*)d_in[4];
  const float* Wv    = (const float*)d_in[5];
  const float* Wo    = (const float*)d_in[6];
  const float* ln1g  = (const float*)d_in[7];
  const float* ln1b  = (const float*)d_in[8];
  const float* ln2g  = (const float*)d_in[9];
  const float* ln2b  = (const float*)d_in[10];
  const unsigned char* maskQ = (const unsigned char*)d_in[11];
  const unsigned char* maskK = (const unsigned char*)d_in[12];

  char* ws = (char*)d_ws;
  const size_t MB = (size_t)1 << 20;
  bf16_t* qin  = (bf16_t*)(ws + 0 * MB);      // 8MB  (reused later as Obf)
  bf16_t* kin  = (bf16_t*)(ws + 8 * MB);      // 8MB
  bf16_t* vin  = (bf16_t*)(ws + 16 * MB);     // 8MB
  bf16_t* wqb  = (bf16_t*)(ws + 24 * MB);     // 2MB each
  bf16_t* wkb  = (bf16_t*)(ws + 26 * MB);
  bf16_t* wvb  = (bf16_t*)(ws + 28 * MB);
  bf16_t* wob  = (bf16_t*)(ws + 30 * MB);
  float*  Qp   = (float*)(ws + 32 * MB);      // 16MB
  float*  Kp   = (float*)(ws + 48 * MB);      // 16MB (reused as h)
  float*  Vp   = (float*)(ws + 64 * MB);      // 16MB (reused as O f32)
  bf16_t* qbf  = (bf16_t*)(ws + 80 * MB);     // 8MB
  bf16_t* kbf  = (bf16_t*)(ws + 88 * MB);     // 8MB
  bf16_t* vT   = (bf16_t*)(ws + 96 * MB);     // 8MB
  float*  Oatt = (float*)(ws + 104 * MB);     // 16MB
  float*  biasK= (float*)(ws + 120 * MB);     // 16KB
  bf16_t* Obf  = qin;   // qin dead after Qp GEMM
  float*  Of   = Vp;    // Vp dead after repack_v
  float*  hbuf = Kp;    // Kp dead after repack_k

  dim3 blk(256);
  // 1) masked bf16 conversion of activations, weights, key bias
  conv_in_kernel<<<4096, blk, 0, stream>>>(Q, maskQ, qin);
  conv_in_kernel<<<4096, blk, 0, stream>>>(K, maskK, kin);
  conv_in_kernel<<<4096, blk, 0, stream>>>(V, maskK, vin);
  conv_w_kernel<<<1024, blk, 0, stream>>>(Wq, wqb);
  conv_w_kernel<<<1024, blk, 0, stream>>>(Wk, wkb);
  conv_w_kernel<<<1024, blk, 0, stream>>>(Wv, wvb);
  conv_w_kernel<<<1024, blk, 0, stream>>>(Wo, wob);
  bias_k_kernel<<<16, blk, 0, stream>>>(maskK, biasK);

  // 2) projections: [4096,1024] x [1024,1024]^T, 256x128 tiles
  dim3 ggrid(4096 / 256, 1024 / 128);
  gemm_bf16_tn<<<ggrid, blk, 0, stream>>>(qin, wqb, Qp, 4096, 1024, 1024);
  gemm_bf16_tn<<<ggrid, blk, 0, stream>>>(kin, wkb, Kp, 4096, 1024, 1024);
  gemm_bf16_tn<<<ggrid, blk, 0, stream>>>(vin, wvb, Vp, 4096, 1024, 1024);

  // 3) head split / transpose to bf16 attention layouts
  repack_qk_kernel<<<4096, blk, 0, stream>>>(Qp, qbf);
  repack_qk_kernel<<<4096, blk, 0, stream>>>(Kp, kbf);
  repack_v_kernel<<<4096, blk, 0, stream>>>(Vp, vT);

  // 4) flash attention
  flash_attn_kernel<<<dim3(8, 16, 4), blk, 0, stream>>>(qbf, kbf, vT, biasK, Oatt);

  // 5) residual + LN1 (masked), fc_o GEMM, GELU residual + LN2 (masked)
  ln1_kernel<<<4096, blk, 0, stream>>>(Qp, Oatt, ln1g, ln1b, maskQ, Of, Obf);
  gemm_bf16_tn<<<ggrid, blk, 0, stream>>>(Obf, wob, hbuf, 4096, 1024, 1024);
  ln2_kernel<<<4096, blk, 0, stream>>>(Of, hbuf, ln2g, ln2b, maskQ, (float*)d_out);
}
